// PAVEModuleV5_10582799417560
// MI455X (gfx1250) — matrix operand order repaired and checked by reading, not compiled, not run on hardware
//
#include <hip/hip_runtime.h>
#include <hip/hip_bf16.h>

typedef unsigned short u16;
typedef __attribute__((ext_vector_type(16))) __bf16 v16bf;
typedef __attribute__((ext_vector_type(8)))  float  v8f;

#define WMMA_BF16(a, b, c) \
  __builtin_amdgcn_wmma_f32_16x16x32_bf16(false, (a), false, (b), (short)0, (c), false, false)

// ---------- dims ----------
#define BB   2
#define MQ   6272      // 32*196 query tokens
#define DQv  896
#define HIDv 512
#define EMBv 1024
#define NKV  2048      // 32*8*8 kv tokens
#define NHh  4
#define HDd  128
#define FFv  1024
#define OUTv 896

__device__ __forceinline__ u16 f2bf(float f) {
  unsigned int u = __float_as_uint(f);
  unsigned int r = (u + 0x7FFFu + ((u >> 16) & 1u)) >> 16;
  return (u16)r;
}
__device__ __forceinline__ float gelu_f(float x) {
  float x3 = x * x * x;
  float t  = tanhf(0.7978845608028654f * (x + 0.044715f * x3));
  return 0.5f * x * (1.0f + t);
}

// ---------- weight convert + transpose: in (K,N) f32 -> out (N,K) bf16 ----------
__global__ void convtrans_kernel(const float* __restrict__ in, u16* __restrict__ out,
                                 int K, int N) {
  long t = (long)blockIdx.x * 256 + threadIdx.x;
  if (t >= (long)K * N) return;
  int n = (int)(t % N), k = (int)(t / N);
  out[(size_t)n * K + k] = f2bf(in[t]);
}

// ---------- straight f32 -> bf16 ----------
__global__ void conv_kernel(const float* __restrict__ in, u16* __restrict__ out, long n) {
  long t = (long)blockIdx.x * 256 + threadIdx.x;
  if (t < n) out[t] = f2bf(in[t]);
}

// ---------- rope cos/sin tables (64 freq channels; halves duplicated) ----------
__global__ void rope_tables_kernel(float* cq, float* sq, float* ck, float* sk) {
  int tid = blockIdx.x * 256 + threadIdx.x;
  const int total = (MQ + NKV) * 64;
  if (tid >= total) return;
  int i = tid & 63, p = tid >> 6;
  int t, hh, ww;
  bool isq = p < MQ;
  if (isq) { t = p / 196; int g = p % 196; hh = g / 14; ww = g % 14; }
  else     { int pk = p - MQ; t = pk / 64; int g = pk % 64; hh = g / 8; ww = g % 8; }
  float pv = (i < 22) ? (float)t : ((i < 43) ? (float)hh : (float)ww);
  float invf = __powf(10000.0f, -(float)(2 * i) / 128.0f);
  float f = pv * invf;
  float c = __cosf(f), s = __sinf(f);
  if (isq) { cq[p * 64 + i] = c; sq[p * 64 + i] = s; }
  else     { int pk = p - MQ; ck[pk * 64 + i] = c; sk[pk * 64 + i] = s; }
}

// ---------- kv = x @ Wm + bm (K=4, scalar) -> bf16 ----------
__global__ void kv_kernel(const float* __restrict__ x, const float* __restrict__ Wm,
                          const float* __restrict__ bm, u16* __restrict__ kv) {
  long t = (long)blockIdx.x * 256 + threadIdx.x;
  const long total = (long)BB * NKV * EMBv;
  if (t >= total) return;
  int e = (int)(t % EMBv);
  long rb = t / EMBv;
  int row = (int)(rb % NKV);
  int b = (int)(rb / NKV);
  const float* xr = x + ((long)b * NKV + row) * 4;
  float acc = bm[e];
  acc += xr[0] * Wm[0 * EMBv + e];
  acc += xr[1] * Wm[1 * EMBv + e];
  acc += xr[2] * Wm[2 * EMBv + e];
  acc += xr[3] * Wm[3 * EMBv + e];
  kv[t] = f2bf(acc);
}

// ---------- generic bf16 WMMA GEMM: C = A(MxK) * Bt(NxK)^T + bias ----------
__global__ __launch_bounds__(256) void gemm_bf16_kernel(
    const u16* __restrict__ A, const u16* __restrict__ Bt,
    const float* __restrict__ bias, float* __restrict__ Cf, u16* __restrict__ Cb,
    int M, int N, int K, long aBatch, long cBatch, int gelu) {
  __shared__ u16 As[128][48];   // +16 pad keeps every b128 LDS access 16B aligned
  __shared__ u16 Bs[128][48];

  int b = blockIdx.z;
  A += (size_t)b * aBatch;
  size_t cOff = (size_t)b * cBatch;

  int tid = threadIdx.x;
  int w  = tid >> 5;
  int ln = tid & 15;
  int hf = (tid >> 4) & 1;
  int wm = w & 3, wn = w >> 2;
  int tileM = blockIdx.y * 128, tileN = blockIdx.x * 128;

  int lrow = tid >> 1;
  int lk   = (tid & 1) * 16;
  const u16* aSrc = A  + (size_t)(tileM + lrow) * K + lk;
  const u16* bSrc = Bt + (size_t)(tileN + lrow) * K + lk;

  v8f acc[2][4];
#pragma unroll
  for (int ms = 0; ms < 2; ms++)
#pragma unroll
    for (int ns = 0; ns < 4; ns++) acc[ms][ns] = (v8f){0,0,0,0,0,0,0,0};

  int nk = K >> 5;
  for (int kt = 0; kt < nk; ++kt) {
    __syncthreads();
    uint4 a0 = *(const uint4*)(aSrc);
    uint4 a1 = *(const uint4*)(aSrc + 8);
    uint4 b0 = *(const uint4*)(bSrc);
    uint4 b1 = *(const uint4*)(bSrc + 8);
    if (kt + 2 < nk) {                        // global_prefetch_b8 two tiles ahead
      __builtin_prefetch(aSrc + 64, 0, 1);
      __builtin_prefetch(bSrc + 64, 0, 1);
    }
    aSrc += 32; bSrc += 32;
    *(uint4*)&As[lrow][lk]     = a0;
    *(uint4*)&As[lrow][lk + 8] = a1;
    *(uint4*)&Bs[lrow][lk]     = b0;
    *(uint4*)&Bs[lrow][lk + 8] = b1;
    __syncthreads();

    v16bf af[2];
#pragma unroll
    for (int ms = 0; ms < 2; ms++) {
      int m = wm * 32 + ms * 16 + ln;
      ((uint4*)&af[ms])[0] = *(const uint4*)&As[m][16 * hf];       // K 0..7 / 16..23
      ((uint4*)&af[ms])[1] = *(const uint4*)&As[m][16 * hf + 8];   // K 8..15 / 24..31
    }
#pragma unroll
    for (int ns = 0; ns < 4; ns++) {
      int n = wn * 64 + ns * 16 + ln;
      v16bf bf2;
      ((uint4*)&bf2)[0] = *(const uint4*)&Bs[n][16 * hf];          // K 0..15 / 16..31
      ((uint4*)&bf2)[1] = *(const uint4*)&Bs[n][16 * hf + 8];
#pragma unroll
      for (int ms = 0; ms < 2; ms++)
        acc[ms][ns] = WMMA_BF16(af[ms], bf2, acc[ms][ns]);
    }
  }

#pragma unroll
  for (int ms = 0; ms < 2; ms++)
#pragma unroll
    for (int ns = 0; ns < 4; ns++)
#pragma unroll
      for (int r = 0; r < 8; r++) {
        int gm = tileM + wm * 32 + ms * 16 + r + 8 * hf;
        int gn = tileN + wn * 64 + ns * 16 + ln;
        float v0 = acc[ms][ns][r];
        if (bias) v0 += bias[gn];
        if (gelu) v0 = gelu_f(v0);
        size_t idx = cOff + (size_t)gm * N + gn;
        if (Cf) Cf[idx] = v0;
        if (Cb) Cb[idx] = f2bf(v0);
      }
}

// ---------- RoPE apply + per-head relayout: f32 [B][M][512] -> bf16 [B][4][M][128] ----------
__global__ void rope_apply_kernel(const float* __restrict__ in, const float* __restrict__ ct,
                                  const float* __restrict__ st, u16* __restrict__ out,
                                  int Mrows) {
  long t = (long)blockIdx.x * 256 + threadIdx.x;
  long total = (long)BB * Mrows * 256;
  if (t >= total) return;
  int i = (int)(t & 63);
  int h = (int)((t >> 6) & 3);
  long rest = t >> 8;
  int m = (int)(rest % Mrows);
  int b = (int)(rest / Mrows);
  const float* row = in + ((size_t)b * Mrows + m) * HIDv + h * HDd;
  float x1 = row[i], x2 = row[64 + i];
  float c = ct[m * 64 + i], s = st[m * 64 + i];
  u16* orow = out + (((size_t)b * NHh + h) * Mrows + m) * HDd;
  orow[i]      = f2bf(x1 * c - x2 * s);
  orow[64 + i] = f2bf(x2 * c + x1 * s);
}

// ---------- v transpose: f32 [B][2048][512] -> bf16 [B][4][128][2048] ----------
__global__ void vt_kernel(const float* __restrict__ vf, u16* __restrict__ out) {
  long t = (long)blockIdx.x * 256 + threadIdx.x;
  const long total = (long)BB * NKV * HIDv;
  if (t >= total) return;
  int c = (int)(t % HIDv);
  long rb = t / HIDv;
  int key = (int)(rb % NKV);
  int b = (int)(rb / NKV);
  int h = c >> 7, d = c & 127;
  out[(((size_t)b * NHh + h) * HDd + d) * NKV + key] = f2bf(vf[t]);
}

// ---------- flash attention: wave = 16 queries x 128 head-dim, K chunks of 32 ----------
__global__ __launch_bounds__(128) void attn_kernel(
    const u16* __restrict__ q,   // [B][4][MQ][128]
    const u16* __restrict__ kT,  // [B][4][2048][128]
    const u16* __restrict__ vT,  // [B][4][128][2048]
    u16* __restrict__ o) {       // [B][MQ][512]
  __shared__ u16 p_lds[4][16][48];

  int tid = threadIdx.x;
  int w  = tid >> 5;
  int ln = tid & 15;
  int hf = (tid >> 4) & 1;
  int h  = blockIdx.y;
  int b  = blockIdx.z;
  int m0 = blockIdx.x * 64 + w * 16;

  const u16* qh = q  + (((size_t)b * NHh + h) * MQ)  * HDd;
  const u16* kh = kT + (((size_t)b * NHh + h) * NKV) * HDd;
  const u16* vh = vT + (((size_t)b * NHh + h) * HDd) * (size_t)NKV;
  u16* op = o + (size_t)b * MQ * HIDv;

  // q fragments (A layout), 4 k-steps of 32 over head dim
  v16bf qf[4];
  const u16* qrow = qh + (size_t)(m0 + ln) * HDd;
#pragma unroll
  for (int ks = 0; ks < 4; ks++) {
    const u16* pA = qrow + ks * 32;
    ((uint4*)&qf[ks])[0] = *(const uint4*)(pA + 16 * hf);
    ((uint4*)&qf[ks])[1] = *(const uint4*)(pA + 16 * hf + 8);
  }

  float m_run[8], l_run[8];
  v8f oacc[8];
#pragma unroll
  for (int r = 0; r < 8; r++) { m_run[r] = -1e30f; l_run[r] = 0.0f; }
#pragma unroll
  for (int ns = 0; ns < 8; ns++) oacc[ns] = (v8f){0,0,0,0,0,0,0,0};

  const float scale = 0.08838834764831845f;  // 1/sqrt(128)

  for (int kc = 0; kc < NKV / 32; ++kc) {
    v8f s0a = (v8f){0,0,0,0,0,0,0,0};
    v8f s1a = (v8f){0,0,0,0,0,0,0,0};
#pragma unroll
    for (int ks = 0; ks < 4; ks++) {
      const u16* kb = kh + (size_t)(kc * 32) * HDd + ks * 32 + 16 * hf;
      v16bf kf0, kf1;
      ((uint4*)&kf0)[0] = *(const uint4*)(kb + (size_t)ln * HDd);
      ((uint4*)&kf0)[1] = *(const uint4*)(kb + (size_t)ln * HDd + 8);
      ((uint4*)&kf1)[0] = *(const uint4*)(kb + (size_t)(16 + ln) * HDd);
      ((uint4*)&kf1)[1] = *(const uint4*)(kb + (size_t)(16 + ln) * HDd + 8);
      s0a = WMMA_BF16(qf[ks], kf0, s0a);
      s1a = WMMA_BF16(qf[ks], kf1, s1a);
    }

    // online softmax per row (rows r+8*hf live in 16-lane halves -> width-16 shfl)
#pragma unroll
    for (int r = 0; r < 8; r++) {
      float s0 = s0a[r] * scale, s1 = s1a[r] * scale;
      float mx = fmaxf(s0, s1);
      mx = fmaxf(mx, __shfl_xor(mx, 1, 16));
      mx = fmaxf(mx, __shfl_xor(mx, 2, 16));
      mx = fmaxf(mx, __shfl_xor(mx, 4, 16));
      mx = fmaxf(mx, __shfl_xor(mx, 8, 16));
      float mnew  = fmaxf(m_run[r], mx);
      float alpha = __expf(m_run[r] - mnew);
      float p0 = __expf(s0 - mnew), p1 = __expf(s1 - mnew);
      float rs = p0 + p1;
      rs += __shfl_xor(rs, 1, 16);
      rs += __shfl_xor(rs, 2, 16);
      rs += __shfl_xor(rs, 4, 16);
      rs += __shfl_xor(rs, 8, 16);
      l_run[r] = l_run[r] * alpha + rs;
      m_run[r] = mnew;
#pragma unroll
      for (int ns = 0; ns < 8; ns++) oacc[ns][r] *= alpha;
      p_lds[w][r + 8 * hf][ln]      = f2bf(p0);
      p_lds[w][r + 8 * hf][16 + ln] = f2bf(p1);
    }

    // P (C layout) -> A layout via per-wave LDS round trip (DS in-order per wave)
    v16bf pf;
    ((uint4*)&pf)[0] = *(const uint4*)&p_lds[w][ln][16 * hf];
    ((uint4*)&pf)[1] = *(const uint4*)&p_lds[w][ln][16 * hf + 8];

#pragma unroll
    for (int ns = 0; ns < 8; ns++) {
      const u16* vb = vh + (size_t)(ns * 16 + ln) * NKV + kc * 32 + 16 * hf;
      v16bf vfg;
      ((uint4*)&vfg)[0] = *(const uint4*)(vb);
      ((uint4*)&vfg)[1] = *(const uint4*)(vb + 8);
      oacc[ns] = WMMA_BF16(pf, vfg, oacc[ns]);
    }
  }

#pragma unroll
  for (int ns = 0; ns < 8; ns++)
#pragma unroll
    for (int r = 0; r < 8; r++) {
      float val = oacc[ns][r] / l_run[r];
      int gm = m0 + r + 8 * hf;
      int gn = h * HDd + ns * 16 + ln;
      op[(size_t)gm * HIDv + gn] = f2bf(val);
    }
}

// ---------- residual add + LayerNorm (1 wave per 512-wide row) ----------
__global__ __launch_bounds__(256) void add_ln_kernel(
    const float* __restrict__ x, const float* __restrict__ res,
    const float* __restrict__ g, const float* __restrict__ bta,
    float* __restrict__ outf, u16* __restrict__ outb, int nrows) {
  int w = threadIdx.x >> 5, ln = threadIdx.x & 31;
  int row = blockIdx.x * 8 + w;
  if (row >= nrows) return;
  const float* xr = x + (size_t)row * HIDv;
  const float* rr = res ? res + (size_t)row * HIDv : nullptr;
  float v[16];
  float s = 0.0f, sq = 0.0f;
#pragma unroll
  for (int i = 0; i < 4; i++) {
    float4 a = *(const float4*)(xr + ln * 16 + i * 4);
    if (rr) {
      float4 b2 = *(const float4*)(rr + ln * 16 + i * 4);
      a.x += b2.x; a.y += b2.y; a.z += b2.z; a.w += b2.w;
    }
    v[i * 4 + 0] = a.x; v[i * 4 + 1] = a.y; v[i * 4 + 2] = a.z; v[i * 4 + 3] = a.w;
    s  += a.x + a.y + a.z + a.w;
    sq += a.x * a.x + a.y * a.y + a.z * a.z + a.w * a.w;
  }
#pragma unroll
  for (int m = 1; m < 32; m <<= 1) { s += __shfl_xor(s, m, 32); sq += __shfl_xor(sq, m, 32); }
  float mean = s * (1.0f / HIDv);
  float var  = sq * (1.0f / HIDv) - mean * mean;
  float rstd = rsqrtf(var + 1e-5f);
#pragma unroll
  for (int i = 0; i < 16; i++) {
    int c = ln * 16 + i;
    float y = (v[i] - mean) * rstd * g[c] + bta[c];
    size_t idx = (size_t)row * HIDv + c;
    if (outf) outf[idx] = y;
    outb[idx] = f2bf(y);
  }
}

// ============================ host launcher ============================
extern "C" void kernel_launch(void* const* d_in, const int* in_sizes, int n_in,
                              void* d_out, int out_size, void* d_ws, size_t ws_size,
                              hipStream_t stream) {
  (void)in_sizes; (void)n_in; (void)out_size; (void)ws_size;
  char* ws = (char*)d_ws;
  size_t off = 0;
  auto alloc = [&](size_t bytes) -> size_t {
    off = (off + 255) & ~(size_t)255;
    size_t o = off; off += bytes; return o;
  };
  // bf16 transposed weights
  size_t wqinT = alloc((size_t)512 * 896 * 2);
  size_t wqT   = alloc((size_t)512 * 512 * 2);
  size_t wkT   = alloc((size_t)512 * 1024 * 2);
  size_t wvT   = alloc((size_t)512 * 1024 * 2);
  size_t woT   = alloc((size_t)512 * 512 * 2);
  size_t w1T   = alloc((size_t)1024 * 512 * 2);
  size_t w2T   = alloc((size_t)512 * 1024 * 2);
  size_t m1T   = alloc((size_t)1024 * 512 * 2);
  size_t m2T   = alloc((size_t)896 * 1024 * 2);
  // rope tables
  size_t cq = alloc((size_t)MQ * 64 * 4);
  size_t sq = alloc((size_t)MQ * 64 * 4);
  size_t ck = alloc((size_t)NKV * 64 * 4);
  size_t sk = alloc((size_t)NKV * 64 * 4);
  // activations
  size_t slow_bf  = alloc((size_t)BB * MQ * DQv * 2);
  size_t qin_f    = alloc((size_t)BB * MQ * HIDv * 4);
  size_t qin_bf   = alloc((size_t)BB * MQ * HIDv * 2);
  size_t kv_bf    = alloc((size_t)BB * NKV * EMBv * 2);
  size_t scr_f    = alloc((size_t)BB * MQ * HIDv * 4);   // qf32 / attn_out / ff2 (disjoint lifetimes)
  size_t kf32     = alloc((size_t)BB * NKV * HIDv * 4);
  size_t vf32     = alloc((size_t)BB * NKV * HIDv * 4);
  size_t q_bf     = alloc((size_t)BB * MQ * HIDv * 2);
  size_t k_bf     = alloc((size_t)BB * NKV * HIDv * 2);
  size_t vt_bf    = alloc((size_t)BB * NKV * HIDv * 2);
  size_t o_bf     = alloc((size_t)BB * MQ * HIDv * 2);
  size_t h1_f     = alloc((size_t)BB * MQ * HIDv * 4);
  size_t h1_bf    = alloc((size_t)BB * MQ * HIDv * 2);
  size_t big_bf   = alloc((size_t)BB * MQ * FFv * 2);    // ff gelu out / mlp gelu out
  size_t h2_bf    = alloc((size_t)BB * MQ * HIDv * 2);

  auto CT = [&](const void* wsrc, size_t o, int K, int N) {
    long n = (long)K * N;
    convtrans_kernel<<<dim3((unsigned)((n + 255) / 256)), 256, 0, stream>>>(
        (const float*)wsrc, (u16*)(ws + o), K, N);
  };
  CT(d_in[3],  wqinT, 896,  512);
  CT(d_in[7],  wqT,   512,  512);
  CT(d_in[9],  wkT,   1024, 512);
  CT(d_in[11], wvT,   1024, 512);
  CT(d_in[13], woT,   512,  512);
  CT(d_in[17], w1T,   512,  1024);
  CT(d_in[19], w2T,   1024, 512);
  CT(d_in[23], m1T,   512,  1024);
  CT(d_in[25], m2T,   1024, 896);

  { long n = (long)BB * MQ * DQv;
    conv_kernel<<<dim3((unsigned)((n + 255) / 256)), 256, 0, stream>>>(
        (const float*)d_in[1], (u16*)(ws + slow_bf), n); }

  { long n = (long)(MQ + NKV) * 64;
    rope_tables_kernel<<<dim3((unsigned)((n + 255) / 256)), 256, 0, stream>>>(
        (float*)(ws + cq), (float*)(ws + sq), (float*)(ws + ck), (float*)(ws + sk)); }

  { long n = (long)BB * NKV * EMBv;
    kv_kernel<<<dim3((unsigned)((n + 255) / 256)), 256, 0, stream>>>(
        (const float*)d_in[0], (const float*)d_in[5], (const float*)d_in[6],
        (u16*)(ws + kv_bf)); }

  auto GEMM = [&](size_t aOff, size_t bOff, const void* bias, size_t cfOff, bool hasCf,
                  size_t cbOff, bool hasCb, int M, int N, int K, int gelu) {
    gemm_bf16_kernel<<<dim3(N / 128, M / 128, BB), 256, 0, stream>>>(
        (const u16*)(ws + aOff), (const u16*)(ws + bOff), (const float*)bias,
        hasCf ? (float*)(ws + cfOff) : nullptr,
        hasCb ? (u16*)(ws + cbOff) : nullptr,
        M, N, K, (long)M * K, (long)M * N, gelu);
  };

  // q_in = slow @ Wq_in + b  -> f32 (residual) + bf16
  GEMM(slow_bf, wqinT, d_in[4], qin_f, true, qin_bf, true, MQ, HIDv, DQv, 0);
  // q/k/v projections -> f32
  GEMM(qin_bf, wqT, d_in[8],  scr_f, true, 0, false, MQ,  HIDv, HIDv, 0);
  GEMM(kv_bf,  wkT, d_in[10], kf32,  true, 0, false, NKV, HIDv, EMBv, 0);
  GEMM(kv_bf,  wvT, d_in[12], vf32,  true, 0, false, NKV, HIDv, EMBv, 0);

  { long n = (long)BB * MQ * 256;
    rope_apply_kernel<<<dim3((unsigned)((n + 255) / 256)), 256, 0, stream>>>(
        (const float*)(ws + scr_f), (const float*)(ws + cq), (const float*)(ws + sq),
        (u16*)(ws + q_bf), MQ); }
  { long n = (long)BB * NKV * 256;
    rope_apply_kernel<<<dim3((unsigned)((n + 255) / 256)), 256, 0, stream>>>(
        (const float*)(ws + kf32), (const float*)(ws + ck), (const float*)(ws + sk),
        (u16*)(ws + k_bf), NKV); }
  { long n = (long)BB * NKV * HIDv;
    vt_kernel<<<dim3((unsigned)((n + 255) / 256)), 256, 0, stream>>>(
        (const float*)(ws + vf32), (u16*)(ws + vt_bf)); }

  attn_kernel<<<dim3(MQ / 64, NHh, BB), 128, 0, stream>>>(
      (const u16*)(ws + q_bf), (const u16*)(ws + k_bf), (const u16*)(ws + vt_bf),
      (u16*)(ws + o_bf));

  // output projection -> f32 (into scratch; qf32 already consumed)
  GEMM(o_bf, woT, d_in[14], scr_f, true, 0, false, MQ, HIDv, HIDv, 0);

  // h1 = LN(q_in + attn_out)
  add_ln_kernel<<<dim3((BB * MQ + 7) / 8), 256, 0, stream>>>(
      (const float*)(ws + qin_f), (const float*)(ws + scr_f),
      (const float*)d_in[15], (const float*)d_in[16],
      (float*)(ws + h1_f), (u16*)(ws + h1_bf), BB * MQ);

  // FFN: gelu(h1 @ W1 + b1) @ W2 + b2
  GEMM(h1_bf,  w1T, d_in[18], 0, false, big_bf, true, MQ, FFv,  HIDv, 1);
  GEMM(big_bf, w2T, d_in[20], scr_f, true, 0, false, MQ, HIDv, FFv,  0);

  // h2 = LN(h1 + ff)
  add_ln_kernel<<<dim3((BB * MQ + 7) / 8), 256, 0, stream>>>(
      (const float*)(ws + h1_f), (const float*)(ws + scr_f),
      (const float*)d_in[21], (const float*)d_in[22],
      nullptr, (u16*)(ws + h2_bf), BB * MQ);

  // output MLP: gelu(h2 @ M1 + bm1) @ M2 + bm2 -> d_out (f32)
  GEMM(h2_bf, m1T, d_in[24], 0, false, big_bf, true, MQ, FFv, HIDv, 1);
  gemm_bf16_kernel<<<dim3(OUTv / 128, MQ / 128, BB), 256, 0, stream>>>(
      (const u16*)(ws + big_bf), (const u16*)(ws + m2T), (const float*)d_in[26],
      (float*)d_out, nullptr, MQ, OUTv, FFv, (long)MQ * FFv, (long)MQ * OUTv, 0);
}
